// MHSA_35107062678092
// MI455X (gfx1250) — matrix operand order repaired and checked by reading, not compiled, not run on hardware
//
#include <hip/hip_runtime.h>

// MHSA (diagonal-attention variant) for gfx1250 / MI455X, bf16 WMMA path.
//
// K0: pack x -> xb (bf16 pairs, [t][kp]), transpose-pack Wqkv -> Wt [n][kp],
//     W0 -> W0t [n][kp]  (one conversion per element, ever).
// K1: qkv GEMM [4096,6144] = x @ Wqkv + bqkv; staging = pure async u32 copies;
//     epilogue scatters bf16 into q/k planes [plane][h][4096][256] + v [4096][2048].
// K2: per (b,h): coeff[t,h] = exp(S[t,t]) / sum_n exp(S[n,t]) via WMMA tiles,
//     never materializing S (diagonal-softmax identity).
// K3: out = (coeff ⊙ v) @ W0 + b0.
//
// ws: qk 32MB + v 16MB + coeff 128KB + xb 2MB + Wt 3MB + W0t 1MB  (~54MB)

typedef __attribute__((ext_vector_type(16))) __bf16        v16bf;
typedef __attribute__((ext_vector_type(8)))  float         v8f;
typedef __attribute__((ext_vector_type(8)))  unsigned int  v8u;

#define TOK     4096
#define DD      256
#define HH      8
#define DH      2048
#define NQKV    6144
#define HSTRIDE (4096u * 256u)       // per-head q/k plane: 1,048,576 elems
#define QKPLANE (8u * HSTRIDE)       // q or k plane: 8,388,608 elems

#define XB_N  524288                 // 4096*256/2   u32 pairs
#define WT_N  786432                 // 256*6144/2
#define W0T_N 262144                 // 2048*256/2

#if __has_builtin(__builtin_amdgcn_global_load_async_to_lds_b32) && \
    __has_builtin(__builtin_amdgcn_s_wait_asynccnt)
#define USE_ASYNC_LDS 1
#else
#define USE_ASYNC_LDS 0
#endif

typedef __attribute__((address_space(1))) int gi32;   // global int*
typedef __attribute__((address_space(3))) int li32;   // LDS int*

// -------- bf16 conversion helpers ------------------------------------------
__device__ __forceinline__ unsigned short f32_to_bf16(float f) {
  return __builtin_bit_cast(unsigned short, (__bf16)f);   // native RNE convert
}

__device__ __forceinline__ unsigned int pack2_bf16(float lo, float hi) {
#if __has_builtin(__builtin_amdgcn_cvt_pk_bf16_f32)
  auto r = __builtin_amdgcn_cvt_pk_bf16_f32(lo, hi);
  return __builtin_bit_cast(unsigned int, r);
#else
  unsigned int a = f32_to_bf16(lo);
  unsigned int b = f32_to_bf16(hi);
  return a | (b << 16);
#endif
}

// -------- async global->LDS u32 copy (falls back to ld+st) -----------------
__device__ __forceinline__ void cp_u32(unsigned int* lds, const unsigned int* g) {
#if USE_ASYNC_LDS
  __builtin_amdgcn_global_load_async_to_lds_b32(
      (gi32*)(unsigned long long)g,
      (li32*)(unsigned int)(unsigned long long)lds, 0, 0);
#else
  *lds = *g;
#endif
}
__device__ __forceinline__ void cp_fence() {
#if USE_ASYNC_LDS
  __builtin_amdgcn_s_wait_asynccnt(0);
#endif
}

__device__ __forceinline__ v8f wmma_bf16(v8u a, v8u b, v8f c) {
  return __builtin_amdgcn_wmma_f32_16x16x32_bf16(
      false, __builtin_bit_cast(v16bf, a),
      false, __builtin_bit_cast(v16bf, b),
      (short)0, c, false, false);
}

// ---------------------------------------------------------------------------
// Kernel 0: one-time pack/transpose of the three GEMM operands to bf16 pairs.
//   xb [4096][128]  = pack(x[t][2kp], x[t][2kp+1])
//   Wt [6144][128]  = pack(Wqkv[2kp][n], Wqkv[2kp+1][n])     (B-LDS layout)
//   W0t[256][1024]  = pack(W0[2kp][n],  W0[2kp+1][n])
// ---------------------------------------------------------------------------
__global__ __launch_bounds__(256) void pack_inputs_kernel(
    const float* __restrict__ x, const float* __restrict__ Wqkv,
    const float* __restrict__ W0, unsigned int* __restrict__ xb,
    unsigned int* __restrict__ Wt, unsigned int* __restrict__ W0t) {
  int i = blockIdx.x * 256 + threadIdx.x;       // [0, XB_N+WT_N+W0T_N)
  if (i < XB_N) {
    const float* s = x + (size_t)i * 2;
    xb[i] = pack2_bf16(s[0], s[1]);
  } else if (i < XB_N + WT_N) {
    int j = i - XB_N;                           // j = kp*6144 + n (read-coalesced)
    int kp = j / NQKV, n = j % NQKV;
    const float* s = Wqkv + (size_t)(2 * kp) * NQKV + n;
    Wt[(size_t)n * 128 + kp] = pack2_bf16(s[0], s[NQKV]);
  } else {
    int j = i - (XB_N + WT_N);                  // j = kp*256 + n
    int kp = j >> 8, n = j & 255;
    const float* s = W0 + (size_t)(2 * kp) * DD + n;
    W0t[(size_t)n * 1024 + kp] = pack2_bf16(s[0], s[DD]);
  }
}

// ---------------------------------------------------------------------------
// Kernel 1: qkv GEMM, pure-async staging, scatter epilogue into bf16 planes.
// block 128x128, 8 waves (2M x 4N), wave = 4x2 grid of 16x16x32 WMMA.
// ---------------------------------------------------------------------------
__global__ __launch_bounds__(256) void qkv_gemm_kernel(
    const unsigned int* __restrict__ xb, const unsigned int* __restrict__ Wt,
    const float* __restrict__ bias, unsigned short* __restrict__ qk,
    unsigned short* __restrict__ vv) {
  __shared__ unsigned int Al[128][17];
  __shared__ unsigned int Bl[128][17];
  const int tid = threadIdx.x, lane = tid & 31, wv = tid >> 5;
  const int m0 = blockIdx.y * 128, n0 = blockIdx.x * 128;
  const int wm = (wv & 1) * 64, wn = (wv >> 1) * 32;
  const int ln = lane & 15, lh = lane >> 4;
  v8f acc[4][2] = {};
  for (int k0 = 0; k0 < DD; k0 += 32) {
    const int kp0 = k0 >> 1;
#pragma unroll
    for (int i = 0; i < 8; ++i) {        // A tile: async u32 copies (k-contig)
      int p = tid + i * 256;
      int row = p >> 4, kp = p & 15;
      cp_u32(&Al[row][kp], xb + (size_t)(m0 + row) * 128 + kp0 + kp);
    }
#pragma unroll
    for (int i = 0; i < 8; ++i) {        // B tile: async u32 copies (pre-transposed)
      int p = tid + i * 256;
      int n = p >> 4, kp = p & 15;
      cp_u32(&Bl[n][kp], Wt + (size_t)(n0 + n) * 128 + kp0 + kp);
    }
    cp_fence();
    __syncthreads();
    v8u bf[2];
#pragma unroll
    for (int nt = 0; nt < 2; ++nt)
#pragma unroll
      for (int r = 0; r < 8; ++r)
        bf[nt][r] = Bl[wn + nt * 16 + ln][lh * 8 + r];
#pragma unroll
    for (int mt = 0; mt < 4; ++mt) {
      v8u af;
#pragma unroll
      for (int r = 0; r < 8; ++r)
        af[r] = Al[wm + mt * 16 + ln][lh * 8 + r];
      acc[mt][0] = wmma_bf16(af, bf[0], acc[mt][0]);
      acc[mt][1] = wmma_bf16(af, bf[1], acc[mt][1]);
    }
    __syncthreads();
  }
  // Epilogue: flat index f = row*6144+col selects plane (q/k/v) and target slot.
#pragma unroll
  for (int mt = 0; mt < 4; ++mt)
#pragma unroll
    for (int nt = 0; nt < 2; ++nt)
#pragma unroll
      for (int r = 0; r < 8; ++r) {
        int grow = m0 + wm + mt * 16 + lh * 8 + r;
        int gcol = n0 + wn + nt * 16 + ln;
        float val = acc[mt][nt][r] + bias[gcol];
        unsigned int f = (unsigned int)grow * (unsigned int)NQKV + (unsigned int)gcol;
        unsigned int plane = f >> 23;           // 0=q, 1=k, 2=v
        unsigned int rr = f & 0x7FFFFFu;
        unsigned short bv = f32_to_bf16(val);
        if (plane < 2) {                        // q/k -> [plane][h][t][d], d-contig
          unsigned int tq = rr >> 11;
          unsigned int d  = (rr >> 3) & 255u;
          unsigned int hh = rr & 7u;
          qk[plane * QKPLANE + hh * HSTRIDE + tq * 256u + d] = bv;
        } else {                                // v -> flat [t][c]
          vv[rr] = bv;
        }
      }
}

// ---------------------------------------------------------------------------
// Kernel 2: coeff[t,h] = exp(S[t,t]) / sum_n exp(S[n,t]); pure-async staging.
// ---------------------------------------------------------------------------
__global__ __launch_bounds__(256) void attn_coeff_kernel(
    const unsigned short* __restrict__ qk, float* __restrict__ coeff) {
  __shared__ unsigned int Ql[128][17];
  __shared__ unsigned int Kl[128][17];
  __shared__ float colsum[128];
  __shared__ float diag[128];
  const int tid = threadIdx.x, lane = tid & 31, wv = tid >> 5;
  const int b = blockIdx.y >> 3, h = blockIdx.y & 7;
  const int c0 = blockIdx.x * 128;
  const int tok0 = b * 1024;
  const unsigned short* qh = qk + (size_t)h * HSTRIDE;
  const unsigned short* kh = qk + (size_t)QKPLANE + (size_t)h * HSTRIDE;
  if (tid < 128) { colsum[tid] = 0.0f; diag[tid] = 0.0f; }
  __syncthreads();
  const int wm = (wv & 1) * 64, wn = (wv >> 1) * 32;
  const int ln = lane & 15, lh = lane >> 4;
  float part[2] = {0.0f, 0.0f};
  for (int nb = 0; nb < 1024; nb += 128) {
    v8f acc[4][2] = {};
    for (int k0 = 0; k0 < DD; k0 += 32) {
#pragma unroll
      for (int i = 0; i < 8; ++i) {      // Q tile: async u32 copies
        int p = tid + i * 256;
        int row = p >> 4, kp = p & 15;
        const unsigned int* s = (const unsigned int*)
            (qh + (size_t)(tok0 + nb + row) * DD + (k0 + kp * 2));
        cp_u32(&Ql[row][kp], s);
      }
#pragma unroll
      for (int i = 0; i < 8; ++i) {      // K tile
        int p = tid + i * 256;
        int col = p >> 4, kp = p & 15;
        const unsigned int* s = (const unsigned int*)
            (kh + (size_t)(tok0 + c0 + col) * DD + (k0 + kp * 2));
        cp_u32(&Kl[col][kp], s);
      }
      cp_fence();
      __syncthreads();
      v8u bf[2];
#pragma unroll
      for (int nt = 0; nt < 2; ++nt)
#pragma unroll
        for (int r = 0; r < 8; ++r)
          bf[nt][r] = Kl[wn + nt * 16 + ln][lh * 8 + r];
#pragma unroll
      for (int mt = 0; mt < 4; ++mt) {
        v8u af;
#pragma unroll
        for (int r = 0; r < 8; ++r)
          af[r] = Ql[wm + mt * 16 + ln][lh * 8 + r];
        acc[mt][0] = wmma_bf16(af, bf[0], acc[mt][0]);
        acc[mt][1] = wmma_bf16(af, bf[1], acc[mt][1]);
      }
      __syncthreads();
    }
#pragma unroll
    for (int mt = 0; mt < 4; ++mt)
#pragma unroll
      for (int nt = 0; nt < 2; ++nt)
#pragma unroll
        for (int r = 0; r < 8; ++r) {
          int lr = wm + mt * 16 + lh * 8 + r;
          int lc = wn + nt * 16 + ln;
          float e = __expf(acc[mt][nt][r] * 0.0625f);   // scale = D^-0.5
          part[nt] += e;
          if (nb + lr == c0 + lc) diag[lc] = e;          // unique writer/col
        }
  }
#pragma unroll
  for (int nt = 0; nt < 2; ++nt) {
    float v = part[nt] + __shfl_xor(part[nt], 16);
    if (lh == 0) atomicAdd(&colsum[wn + nt * 16 + ln], v);  // ds_add_f32
  }
  __syncthreads();
  if (tid < 128) {
    int t = tok0 + c0 + tid;
    coeff[(size_t)t * HH + h] = diag[tid] / colsum[tid];
  }
}

// ---------------------------------------------------------------------------
// Kernel 3: out[4096,256] = (coeff ⊙ v_bf16)[4096,2048] @ W0[2048,256] + b0
// ---------------------------------------------------------------------------
__global__ __launch_bounds__(256) void out_gemm_kernel(
    const unsigned short* __restrict__ vv, const float* __restrict__ coeff,
    const unsigned int* __restrict__ W0t, const float* __restrict__ b0,
    float* __restrict__ out) {
  __shared__ unsigned int Al[128][17];
  __shared__ unsigned int Bl[128][17];
  __shared__ float cf[128][8];
  const int tid = threadIdx.x, lane = tid & 31, wv = tid >> 5;
  const int m0 = blockIdx.y * 128, n0 = blockIdx.x * 128;
  for (int i = tid; i < 1024; i += 256)
    cf[i >> 3][i & 7] = coeff[(size_t)m0 * HH + i];
  __syncthreads();
  const int wm = (wv & 1) * 64, wn = (wv >> 1) * 32;
  const int ln = lane & 15, lh = lane >> 4;
  v8f acc[4][2] = {};
  for (int k0 = 0; k0 < DH; k0 += 32) {
    const int kp0 = k0 >> 1;
#pragma unroll
    for (int i = 0; i < 8; ++i) {        // A = coeff * v (bf16 pair -> scale -> repack)
      int p = tid + i * 256;
      int row = p >> 4, kp = p & 15;
      int kk = k0 + kp * 2;
      unsigned int u = *(const unsigned int*)(vv + (size_t)(m0 + row) * DH + kk);
      float lo = __builtin_bit_cast(float, u << 16);
      float hi = __builtin_bit_cast(float, u & 0xFFFF0000u);
      Al[row][kp] = pack2_bf16(lo * cf[row][kk & 7], hi * cf[row][(kk + 1) & 7]);
    }
#pragma unroll
    for (int i = 0; i < 8; ++i) {        // B tile: async u32 copies (pre-transposed)
      int p = tid + i * 256;
      int n = p >> 4, kp = p & 15;
      cp_u32(&Bl[n][kp], W0t + (size_t)(n0 + n) * 1024 + kp0 + kp);
    }
    cp_fence();
    __syncthreads();
    v8u bf[2];
#pragma unroll
    for (int nt = 0; nt < 2; ++nt)
#pragma unroll
      for (int r = 0; r < 8; ++r)
        bf[nt][r] = Bl[wn + nt * 16 + ln][lh * 8 + r];
#pragma unroll
    for (int mt = 0; mt < 4; ++mt) {
      v8u af;
#pragma unroll
      for (int r = 0; r < 8; ++r)
        af[r] = Al[wm + mt * 16 + ln][lh * 8 + r];
      acc[mt][0] = wmma_bf16(af, bf[0], acc[mt][0]);
      acc[mt][1] = wmma_bf16(af, bf[1], acc[mt][1]);
    }
    __syncthreads();
  }
#pragma unroll
  for (int mt = 0; mt < 4; ++mt)
#pragma unroll
    for (int nt = 0; nt < 2; ++nt)
#pragma unroll
      for (int r = 0; r < 8; ++r) {
        int grow = m0 + wm + mt * 16 + lh * 8 + r;
        int gcol = n0 + wn + nt * 16 + ln;
        out[(size_t)grow * DD + gcol] = acc[mt][nt][r] + b0[gcol];
      }
}

extern "C" void kernel_launch(void* const* d_in, const int* in_sizes, int n_in,
                              void* d_out, int out_size, void* d_ws, size_t ws_size,
                              hipStream_t stream) {
  const float* x    = (const float*)d_in[0];
  const float* Wqkv = (const float*)d_in[1];
  const float* bqkv = (const float*)d_in[2];
  const float* W0   = (const float*)d_in[3];
  const float* b0   = (const float*)d_in[4];
  float* out = (float*)d_out;

  unsigned short* qk  = (unsigned short*)d_ws;             // 2*QKPLANE bf16 (32MB)
  unsigned short* vv  = qk + 2 * (size_t)QKPLANE;          // TOK*DH bf16   (16MB)
  float*         coef = (float*)(vv + (size_t)TOK * DH);   // 32768 f32
  unsigned int*  xb   = (unsigned int*)(coef + TOK * HH);  // XB_N  u32
  unsigned int*  Wt   = xb + XB_N;                         // WT_N  u32
  unsigned int*  W0t  = Wt + WT_N;                         // W0T_N u32

  pack_inputs_kernel<<<dim3((XB_N + WT_N + W0T_N) / 256), 256, 0, stream>>>(
      x, Wqkv, W0, xb, Wt, W0t);
  qkv_gemm_kernel  <<<dim3(48, 32), 256, 0, stream>>>(xb, Wt, bqkv, qk, vv);
  attn_coeff_kernel<<<dim3(8, 32),  256, 0, stream>>>(qk, coef);
  out_gemm_kernel  <<<dim3(2, 32),  256, 0, stream>>>(vv, coef, W0t, b0, out);

  (void)in_sizes; (void)n_in; (void)out_size; (void)ws_size;
}
